// DCNv3_3D_79860621902194
// MI455X (gfx1250) — matrix operand order repaired and checked by reading, not compiled, not run on hardware
//
#include <hip/hip_runtime.h>
#include <hip/hip_bf16.h>
#include <math.h>

typedef __attribute__((ext_vector_type(16))) _Float16 v16h;
typedef __attribute__((ext_vector_type(8)))  float    v8f;

#define DHW   32768          // 32*32*32 voxels
#define CCH   64
#define NTOT  (DHW * CCH)    // 2097152 elements
#define DPAD  34
#define PADVOL (DPAD * DPAD * DPAD * CCH)   // 2515456 floats

// gfx1250 async global->LDS path, guarded so either toolchain compiles.
#if defined(__HIP_DEVICE_COMPILE__) && \
    __has_builtin(__builtin_amdgcn_global_load_async_to_lds_b128) && \
    __has_builtin(__builtin_amdgcn_s_wait_asynccnt)
#define USE_ASYNC_LDS 1
#else
#define USE_ASYNC_LDS 0
#endif

#if USE_ASYNC_LDS
// Builtin signature (from hipcc diagnostic): (v4i AS1*, v4i AS3*, Imm int, Imm int)
typedef int v4i_ __attribute__((vector_size(16)));
typedef __attribute__((address_space(1))) v4i_ gv4i_t;
typedef __attribute__((address_space(3))) v4i_ lv4i_t;
#define ASYNC_G2L_B128(gsrc, ldst) \
  __builtin_amdgcn_global_load_async_to_lds_b128( \
      (gv4i_t*)(void*)(gsrc), (lv4i_t*)(void*)(ldst), 0, 0)
#endif

// ---------------------------------------------------------------------------
// 1. zero the padded sample volume
// ---------------------------------------------------------------------------
__global__ void k_init(float* __restrict__ pad) {
  int i = blockIdx.x * blockDim.x + threadIdx.x;
  if (i < PADVOL) pad[i] = 0.f;
}

// ---------------------------------------------------------------------------
// 2. depthwise 3x3x3 conv (zero pad) + per-block partial sums for GroupNorm(1)
// ---------------------------------------------------------------------------
__global__ void k_dwconv(const float* __restrict__ x, const float* __restrict__ dww,
                         float* __restrict__ xc,
                         float* __restrict__ psum, float* __restrict__ psq) {
  int idx = blockIdx.x * blockDim.x + threadIdx.x;   // grid covers NTOT exactly
  int c = idx >> 15;
  int v = idx & 32767;
  int d = v >> 10, h = (v >> 5) & 31, w = v & 31;
  const float* xb = x + ((size_t)c << 15);
  const float* wt = dww + c * 27;
  float val = 0.f;
  #pragma unroll
  for (int kz = 0; kz < 3; ++kz) {
    int dz = d + kz - 1;
    if ((unsigned)dz >= 32u) continue;
    #pragma unroll
    for (int ky = 0; ky < 3; ++ky) {
      int dy = h + ky - 1;
      if ((unsigned)dy >= 32u) continue;
      #pragma unroll
      for (int kx = 0; kx < 3; ++kx) {
        int dx = w + kx - 1;
        if ((unsigned)dx >= 32u) continue;
        val += wt[(kz * 3 + ky) * 3 + kx] * xb[((dz * 32) + dy) * 32 + dx];
      }
    }
  }
  xc[idx] = val;

  __shared__ float s1[256], s2[256];
  int t = threadIdx.x;
  s1[t] = val; s2[t] = val * val;
  __syncthreads();
  for (int s = 128; s > 0; s >>= 1) {
    if (t < s) { s1[t] += s1[t + s]; s2[t] += s2[t + s]; }
    __syncthreads();
  }
  if (t == 0) { psum[blockIdx.x] = s1[0]; psq[blockIdx.x] = s2[0]; }
}

// ---------------------------------------------------------------------------
// 3. deterministic final reduction -> mean, inv_std
// ---------------------------------------------------------------------------
__global__ void k_finstats(const float* __restrict__ psum, const float* __restrict__ psq,
                           float* __restrict__ stats) {
  __shared__ float s1[256], s2[256];
  int t = threadIdx.x;
  float a = 0.f, b = 0.f;
  for (int i = t; i < 8192; i += 256) { a += psum[i]; b += psq[i]; }
  s1[t] = a; s2[t] = b;
  __syncthreads();
  for (int s = 128; s > 0; s >>= 1) {
    if (t < s) { s1[t] += s1[t + s]; s2[t] += s2[t + s]; }
    __syncthreads();
  }
  if (t == 0) {
    float inv = 1.f / (float)NTOT;
    float mu  = s1[0] * inv;
    float var = s2[0] * inv - mu * mu;
    stats[0] = mu;
    stats[1] = rsqrtf(var + 1e-5f);
  }
}

// ---------------------------------------------------------------------------
// 4. GroupNorm + GELU(tanh) -> f16 x1 [v,64]; also f16 transpose of x [v,64]
// ---------------------------------------------------------------------------
__global__ void k_normpack(const float* __restrict__ xc, const float* __restrict__ x,
                           const float* __restrict__ gnw, const float* __restrict__ gnb,
                           const float* __restrict__ stats,
                           _Float16* __restrict__ x1h, _Float16* __restrict__ xth) {
  int idx = blockIdx.x * blockDim.x + threadIdx.x;
  int c = idx >> 15;
  int v = idx & 32767;
  float mu = stats[0], is = stats[1];
  float xn = (xc[idx] - mu) * is * gnw[c] + gnb[c];
  float g  = 0.5f * xn * (1.f + tanhf(0.7978845608f * (xn + 0.044715f * xn * xn * xn)));
  x1h[(size_t)v * 64 + c] = (_Float16)g;
  xth[(size_t)v * 64 + c] = (_Float16)x[idx];
}

// ---------------------------------------------------------------------------
// 5. pack weights to f16 (off_w | mask_w padded to 224 cols), biases to 224
// ---------------------------------------------------------------------------
__global__ void k_convw(const float* __restrict__ off_w, const float* __restrict__ mask_w,
                        const float* __restrict__ inp_w, const float* __restrict__ out_w,
                        const float* __restrict__ off_b, const float* __restrict__ mask_b,
                        _Float16* __restrict__ Bom, _Float16* __restrict__ Binp,
                        _Float16* __restrict__ Bout, float* __restrict__ bias224) {
  int i = blockIdx.x * blockDim.x + threadIdx.x;
  if (i < 64 * 224) {
    int k = i / 224, n = i % 224;
    float val = (n < 162) ? off_w[k * 162 + n]
                          : ((n < 216) ? mask_w[k * 54 + (n - 162)] : 0.f);
    Bom[i] = (_Float16)val;
  } else if (i < 64 * 224 + 4096) {
    int j = i - 64 * 224;
    Binp[j] = (_Float16)inp_w[j];
  } else if (i < 64 * 224 + 8192) {
    int j = i - 64 * 224 - 4096;
    Bout[j] = (_Float16)out_w[j];
  } else if (i < 64 * 224 + 8192 + 224) {
    int j = i - 64 * 224 - 8192;
    bias224[j] = (j < 162) ? off_b[j] : ((j < 216) ? mask_b[j - 162] : 0.f);
  }
}

// ---------------------------------------------------------------------------
// WMMA GEMM: C[M x NCOL] = A[M x 64](f16) * B[64 x NCOL](f16) + bias, f32 acc.
// Block = 8 waves sharing one 16-col B panel staged in LDS (async on gfx1250).
// Each wave does 4 consecutive 16x16 tiles along M (64 rows), 2 WMMA k-steps.
// NCOL/LDC are compile-time so all store addressing folds into immediate
// offsets on one base (no per-store 64-bit add chains).
// MODE 0: row-major store (LDC). MODE 1: scatter into padded [34,34,34,64].
// VGPR layouts per CDNA5 ISA 7.12.2 (wave32).
// ---------------------------------------------------------------------------
template <int MODE, int NCOL, int LDC>
__global__ void k_gemm(const _Float16* __restrict__ A, const _Float16* __restrict__ B,
                       const float* __restrict__ bias, float* __restrict__ C) {
  __shared__ _Float16 Bs[64 * 16];        // one K=64 x N=16 panel, 2 KB
  constexpr int ntiles = NCOL >> 4;
  int nt = blockIdx.x % ntiles;
  int n0 = nt << 4;

  // ---- stage B panel into LDS (rows are 16B aligned for all NCOL used) ----
#if USE_ASYNC_LDS
  if (threadIdx.x < 128) {
    int rowk  = threadIdx.x >> 1;
    int halfp = threadIdx.x & 1;
    const _Float16* gsrc = B + (size_t)rowk * NCOL + n0 + halfp * 8;
    ASYNC_G2L_B128(gsrc, &Bs[rowk * 16 + halfp * 8]);
  }
  __builtin_amdgcn_s_wait_asynccnt(0);
  __syncthreads();
#else
  for (int i = threadIdx.x; i < 64 * 16; i += 256) {
    int rowk = i >> 4, n = i & 15;
    Bs[i] = B[(size_t)rowk * NCOL + n0 + n];
  }
  __syncthreads();
#endif

  int lane  = threadIdx.x & 31;
  int nB    = lane & 15;
  int koffB = (lane & 16) ? 8 : 0;        // B: lanes16-31 hold K+8 block
  int koffA = (lane & 16) ? 8 : 0;        // A: lanes16-31 hold K+8 pairs

  v16h b0, b1;
  {
    const _Float16* Bp0 = Bs + koffB * 16 + nB;
    const _Float16* Bp1 = Bs + (32 + koffB) * 16 + nB;
    #pragma unroll
    for (int j = 0; j < 16; ++j) b0[j] = Bp0[(j + (j & 8)) * 16];
    #pragma unroll
    for (int j = 0; j < 16; ++j) b1[j] = Bp1[(j + (j & 8)) * 16];
  }
  float bv = bias[n0 + nB];

  int mg = (blockIdx.x / ntiles) * 8 + (threadIdx.x >> 5);  // 64-row group

  for (int mi = 0; mi < 4; ++mi) {
    int m0  = (mg << 6) + (mi << 4);
    int row = m0 + (lane & 15);
    const _Float16* Ar = A + (size_t)row * 64;
    if (mi < 3) __builtin_prefetch(Ar + 16 * 64, 0, 3);   // next A tile
    v16h a0, a1;
    #pragma unroll
    for (int i = 0; i < 8; ++i) {
      int kl = ((i < 4) ? (2 * i) : (8 + 2 * i)) + koffA; // {0,2,4,6,16,18,20,22}+koffA
      a0[2 * i]     = Ar[kl];
      a0[2 * i + 1] = Ar[kl + 1];
      a1[2 * i]     = Ar[32 + kl];
      a1[2 * i + 1] = Ar[32 + kl + 1];
    }
    v8f c = {0.f, 0.f, 0.f, 0.f, 0.f, 0.f, 0.f, 0.f};
    c = __builtin_amdgcn_wmma_f32_16x16x32_f16(false, a0, false, b0, (short)0, c, false, false);
    c = __builtin_amdgcn_wmma_f32_16x16x32_f16(false, a1, false, b1, (short)0, c, false, false);

    int rbase = m0 + ((lane >> 4) << 3);  // D: lanes16-31 -> rows M+8..M+15
    if (MODE == 0) {
      float* Cp = C + (size_t)rbase * LDC + n0 + nB;
      #pragma unroll
      for (int r = 0; r < 8; ++r) Cp[r * LDC] = c[r] + bv;  // imm offsets
    } else {
      // rbase..rbase+7 never cross a 32-row (W) boundary: hoist d/h/w math.
      int d = rbase >> 10, h = (rbase >> 5) & 31, wb = rbase & 31;
      float* Cp = C + ((size_t)((d + 1) * DPAD + (h + 1)) * DPAD + (wb + 1)) * 64 + n0 + nB;
      #pragma unroll
      for (int r = 0; r < 8; ++r) Cp[r * 64] = c[r] + bv;   // imm offsets
    }
  }
}

// ---------------------------------------------------------------------------
// 7. softmax over the 27 kernel points per (voxel, group), in place
// ---------------------------------------------------------------------------
__global__ void k_softmax(float* __restrict__ offmask) {
  int t = blockIdx.x * blockDim.x + threadIdx.x;
  if (t >= DHW * 2) return;
  int v = t >> 1, g = t & 1;
  float* p = offmask + (size_t)v * 224 + 162 + g * 27;
  float mx = p[0];
  #pragma unroll
  for (int i = 1; i < 27; ++i) mx = fmaxf(mx, p[i]);
  float e[27], s = 0.f;
  #pragma unroll
  for (int i = 0; i < 27; ++i) { e[i] = expf(p[i] - mx); s += e[i]; }
  float inv = 1.f / s;
  #pragma unroll
  for (int i = 0; i < 27; ++i) p[i] = e[i] * inv;
}

// ---------------------------------------------------------------------------
// 9. deformable trilinear sampling. One wave per (voxel, group); lanes are the
//    32 channels of the group -> every corner fetch is one contiguous 128B
//    load from the channel-last padded volume. The 4 offmask rows a block
//    needs (4 voxels x 2 groups = 8 waves) are staged in LDS (async-capable).
// ---------------------------------------------------------------------------
__global__ void k_sample(const float* __restrict__ offmask, const float* __restrict__ pad,
                         _Float16* __restrict__ y) {
  __shared__ float smem[4 * 224];         // 4 contiguous offmask rows, 3584 B
  int vbase = blockIdx.x * 4;
#if USE_ASYNC_LDS
  if (threadIdx.x < 224) {
    const float* g = offmask + (size_t)vbase * 224 + threadIdx.x * 4;
    ASYNC_G2L_B128(g, &smem[threadIdx.x * 4]);
  }
  __builtin_amdgcn_s_wait_asynccnt(0);
  __syncthreads();
#else
  for (int i = threadIdx.x; i < 4 * 224; i += 256)
    smem[i] = offmask[(size_t)vbase * 224 + i];
  __syncthreads();
#endif

  int lane = threadIdx.x & 31;
  int wl   = threadIdx.x >> 5;            // 0..7 local wave
  int v = vbase + (wl >> 1);
  int g = wl & 1;
  int d = v >> 10, h = (v >> 5) & 31, w = v & 31;
  const float* om = smem + (wl >> 1) * 224;
  const float* of = om + g * 81;          // [p][xyz]
  const float* mk = om + 162 + g * 27;    // softmaxed mask
  int ch = g * 32 + lane;
  float wf = (float)w, hf = (float)h, df = (float)d;
  float acc = 0.f;

  #pragma unroll 1
  for (int p = 0; p < 27; ++p) {
    int kx = p % 3 - 1, ky = (p / 3) % 3 - 1, kz = p / 9 - 1;
    float ox = of[p * 3 + 0], oy = of[p * 3 + 1], oz = of[p * 3 + 2];
    float mval = mk[p];
    // ix = loc01*Wp - 0.5 simplifies to integer-grid form:
    float ix = wf + 1.f + (float)kx + 0.25f * ox;   // axis_scale.x * cur_scale = 0.25
    float iy = hf + 1.f + (float)ky + 0.5f  * oy;   // 1.0 * 0.5
    float iz = df + 1.f + (float)kz + 0.5f  * oz;
    float fx0 = floorf(ix), fy0 = floorf(iy), fz0 = floorf(iz);
    int x0 = (int)fx0, y0 = (int)fy0, z0 = (int)fz0;
    float fx = ix - fx0, fy = iy - fy0, fz = iz - fz0;
    float res = 0.f;
    #pragma unroll
    for (int dz2 = 0; dz2 < 2; ++dz2) {
      int cz = z0 + dz2;
      if ((unsigned)cz >= (unsigned)DPAD) continue;
      float wz = dz2 ? fz : 1.f - fz;
      #pragma unroll
      for (int dy2 = 0; dy2 < 2; ++dy2) {
        int cy = y0 + dy2;
        if ((unsigned)cy >= (unsigned)DPAD) continue;
        float wy = dy2 ? fy : 1.f - fy;
        #pragma unroll
        for (int dx2 = 0; dx2 < 2; ++dx2) {
          int cx = x0 + dx2;
          if ((unsigned)cx >= (unsigned)DPAD) continue;
          float wx = dx2 ? fx : 1.f - fx;
          res += wz * wy * wx * pad[((size_t)(cz * DPAD + cy) * DPAD + cx) * 64 + ch];
        }
      }
    }
    acc += mval * res;
  }
  y[(size_t)v * 64 + ch] = (_Float16)acc;
}

// ---------------------------------------------------------------------------
extern "C" void kernel_launch(void* const* d_in, const int* in_sizes, int n_in,
                              void* d_out, int out_size, void* d_ws, size_t ws_size,
                              hipStream_t stream) {
  const float* x      = (const float*)d_in[0];
  const float* dw_w   = (const float*)d_in[1];
  const float* gn_w   = (const float*)d_in[2];
  const float* gn_b   = (const float*)d_in[3];
  const float* inp_w  = (const float*)d_in[4];
  const float* inp_b  = (const float*)d_in[5];
  const float* off_w  = (const float*)d_in[6];
  const float* off_b  = (const float*)d_in[7];
  const float* mask_w = (const float*)d_in[8];
  const float* mask_b = (const float*)d_in[9];
  const float* out_w  = (const float*)d_in[10];
  const float* out_b  = (const float*)d_in[11];

  char* ws = (char*)d_ws;
  size_t off = 0;
  auto take = [&](size_t bytes) -> void* {
    void* p = ws + off;
    off = (off + bytes + 255) & ~(size_t)255;
    return p;
  };
  float*    xc      = (float*)   take(sizeof(float) * NTOT);
  float*    psum    = (float*)   take(sizeof(float) * 8192);
  float*    psq     = (float*)   take(sizeof(float) * 8192);
  float*    stats   = (float*)   take(sizeof(float) * 4);
  _Float16* x1h     = (_Float16*)take(sizeof(_Float16) * NTOT);
  _Float16* xth     = (_Float16*)take(sizeof(_Float16) * NTOT);
  _Float16* Bom     = (_Float16*)take(sizeof(_Float16) * 64 * 224);
  _Float16* Binp    = (_Float16*)take(sizeof(_Float16) * 64 * 64);
  _Float16* Bout    = (_Float16*)take(sizeof(_Float16) * 64 * 64);
  float*    bias224 = (float*)   take(sizeof(float) * 224);
  float*    offmask = (float*)   take(sizeof(float) * (size_t)DHW * 224);
  float*    xpad    = (float*)   take(sizeof(float) * PADVOL);
  _Float16* yh      = (_Float16*)take(sizeof(_Float16) * NTOT);

  k_init<<<(PADVOL + 255) / 256, 256, 0, stream>>>(xpad);
  k_dwconv<<<NTOT / 256, 256, 0, stream>>>(x, dw_w, xc, psum, psq);
  k_finstats<<<1, 256, 0, stream>>>(psum, psq, stats);
  k_normpack<<<NTOT / 256, 256, 0, stream>>>(xc, x, gn_w, gn_b, stats, x1h, xth);
  k_convw<<<(64 * 224 + 8192 + 224 + 255) / 256, 256, 0, stream>>>(
      off_w, mask_w, inp_w, out_w, off_b, mask_b, Bom, Binp, Bout, bias224);

  {  // offsets + mask logits: [32768 x 64] @ [64 x 224]
    int blocks = (224 / 16) * (DHW / 64 / 8);   // ntiles * mgroups/8 = 14*64
    k_gemm<0, 224, 224><<<blocks, 256, 0, stream>>>(x1h, Bom, bias224, offmask);
  }
  k_softmax<<<(DHW * 2 + 255) / 256, 256, 0, stream>>>(offmask);
  {  // xp = x^T @ inp_w, scattered into padded channel-last volume
    int blocks = (64 / 16) * (DHW / 64 / 8);    // 4*64
    k_gemm<1, 64, 0><<<blocks, 256, 0, stream>>>(xth, Binp, inp_b, xpad);
  }
  k_sample<<<(DHW * 2) / 8, 256, 0, stream>>>(offmask, xpad, yh);
  {  // out = y @ out_w + out_b -> d_out [32768 x 64] f32
    int blocks = (64 / 16) * (DHW / 64 / 8);    // 4*64
    k_gemm<0, 64, 64><<<blocks, 256, 0, stream>>>(yh, Bout, out_b, (float*)d_out);
  }
}
